// MultiHeadAttention_11699490914448
// MI455X (gfx1250) — matrix operand order repaired and checked by reading, not compiled, not run on hardware
//
#include <hip/hip_runtime.h>
#include <hip/hip_bf16.h>

#define HEADS 16
#define DIM   128
#define SEQ   2048
#define NEGV  (-1e9f)
#define SCALE 0.02209708691207961f   /* 1/sqrt(2048) */

typedef __attribute__((ext_vector_type(16))) _Float16     v16h;
typedef __attribute__((ext_vector_type(8)))  _Float16     v8h;
typedef __attribute__((ext_vector_type(8)))  float        v8f;
typedef __attribute__((ext_vector_type(4)))  unsigned int u32x4;
typedef __attribute__((ext_vector_type(4)))  int          i32x4;
typedef __attribute__((ext_vector_type(8)))  int          i32x8;

#if __has_builtin(__builtin_amdgcn_tensor_load_to_lds)
#define USE_TDM 1
#else
#define USE_TDM 0
#endif

// ---------------------------------------------------------------------------
// Fragment loaders (ISA 7.12.2 layouts, wave32)
// ---------------------------------------------------------------------------

// A-matrix 16x32 f16: lanes 0-15 row M=lane, K=0..7 (v0-3) & 16..23 (v4-7);
// lanes 16-31 row M=lane-16, K=8..15 & 24..31.
__device__ inline v16h load_fragA(const _Float16* base, int ld, int lane) {
  int r  = lane & 15;
  int kb = (lane >> 4) << 3;                 // 0 or 8
  const _Float16* p = base + r * ld + kb;
  v8h a0 = *(const v8h*)(p);                 // 16B -> global_load_b128 / ds_load_b128
  v8h a1 = *(const v8h*)(p + 16);
  v16h o;
#pragma unroll
  for (int t = 0; t < 8; ++t) { o[t] = a0[t]; o[8 + t] = a1[t]; }
  return o;
}

// B-matrix 32x16 f16 loaded from B^T stored row-major (N rows x K cols):
// lanes 0-15 col N=lane, K=0..7 & 16..23; lanes 16-31 col N=lane-16, K=8..15 & 24..31.
__device__ inline v16h load_fragB(const _Float16* base, int ld, int lane) {
  int n  = lane & 15;
  int kb = (lane >> 4) << 3;                 // 0 or 8
  const _Float16* p = base + n * ld + kb;
  v8h b0 = *(const v8h*)(p);
  v8h b1 = *(const v8h*)(p + 16);
  v16h o;
#pragma unroll
  for (int t = 0; t < 8; ++t) { o[t] = b0[t]; o[8 + t] = b1[t]; }
  return o;
}

// A fragment from an f32 source with on-the-fly f16 conversion.
__device__ inline v16h load_fragA_f32(const float* base, int ld, int lane) {
  int r  = lane & 15;
  int kb = (lane >> 4) << 3;
  const float* p = base + r * ld + kb;
  v16h o;
#pragma unroll
  for (int t = 0; t < 8; ++t) {
    o[t]     = (_Float16)p[t];
    o[8 + t] = (_Float16)p[16 + t];
  }
  return o;
}

__device__ inline v8f wmma16(v16h a, v16h b, v8f c) {
  return __builtin_amdgcn_wmma_f32_16x16x32_f16(false, a, false, b,
                                                (short)0, c, false, false);
}

// Cross-lane reductions over the 16 lanes of a half-wave (xor < 16 keeps half).
__device__ inline float red_max16(float x) {
  x = fmaxf(x, __uint_as_float(__builtin_amdgcn_ds_swizzle(__float_as_uint(x), 0x041F)));
  x = fmaxf(x, __uint_as_float(__builtin_amdgcn_ds_swizzle(__float_as_uint(x), 0x081F)));
  x = fmaxf(x, __uint_as_float(__builtin_amdgcn_ds_swizzle(__float_as_uint(x), 0x101F)));
  x = fmaxf(x, __uint_as_float(__builtin_amdgcn_ds_swizzle(__float_as_uint(x), 0x201F)));
  return x;
}
__device__ inline float red_sum16(float x) {
  x += __uint_as_float(__builtin_amdgcn_ds_swizzle(__float_as_uint(x), 0x041F));
  x += __uint_as_float(__builtin_amdgcn_ds_swizzle(__float_as_uint(x), 0x081F));
  x += __uint_as_float(__builtin_amdgcn_ds_swizzle(__float_as_uint(x), 0x101F));
  x += __uint_as_float(__builtin_amdgcn_ds_swizzle(__float_as_uint(x), 0x201F));
  return x;
}

#if USE_TDM
// ---------------------------------------------------------------------------
// Tensor Data Mover: DMA a (up to) 2D f16 tile from global into LDS.
// tile_d1 == 0 -> 1D tile of tile_d0 contiguous elements.
// Descriptor bitfields per CDNA5 ISA ch.8 (D# groups 0/1; groups 2/3 unused).
// ---------------------------------------------------------------------------
__device__ inline void tdm_load_f16(_Float16* lds_dst, const _Float16* gsrc,
                                    unsigned tensor_d0, unsigned tensor_d1,
                                    unsigned tile_d0, unsigned tile_d1,
                                    unsigned long long row_stride_elems) {
  unsigned long long ga = (unsigned long long)(uintptr_t)gsrc;
  unsigned lds = (unsigned)(uintptr_t)lds_dst;   // low 32b of LDS aperture = offset

  u32x4 g0;
  g0[0] = 1u;                                    // count=1 (valid user descriptor)
  g0[1] = lds;                                   // lds_addr
  g0[2] = (unsigned)(ga & 0xffffffffu);          // global_addr[31:0]
  g0[3] = (unsigned)((ga >> 32) & 0x01ffffffu)   // global_addr[56:32]
        | (2u << 30);                            // type = 2 ("image")

  i32x8 g1;
  g1[0] = (int)(1u << 16);                       // workgroup_mask=0, data_size=2B
  g1[1] = (int)((tensor_d0 & 0xffffu) << 16);    // tensor_dim0[15:0]
  g1[2] = (int)((tensor_d0 >> 16) | ((tensor_d1 & 0xffffu) << 16));
  g1[3] = (int)((tensor_d1 >> 16) | ((tile_d0 & 0xffffu) << 16));
  g1[4] = (int)(tile_d1 & 0xffffu);              // tile_dim1 (tile_dim2=0)
  g1[5] = (int)(row_stride_elems & 0xffffffffull);        // dim0_stride[31:0]
  g1[6] = (int)((row_stride_elems >> 32) & 0xffffull);    // dim0_stride[47:32]
  g1[7] = 0;                                     // dim1_stride unused

  i32x4 gz;  gz[0] = gz[1] = gz[2] = gz[3] = 0;
#if defined(__clang_major__) && (__clang_major__ >= 23)
  i32x8 gz8;
#pragma unroll
  for (int t = 0; t < 8; ++t) gz8[t] = 0;
  __builtin_amdgcn_tensor_load_to_lds(g0, g1, gz, gz, gz8, 0);
#else
  __builtin_amdgcn_tensor_load_to_lds(g0, g1, gz, gz, 0);
#endif
}

__device__ inline void tdm_wait0() {
#if __has_builtin(__builtin_amdgcn_s_wait_tensorcnt)
  __builtin_amdgcn_s_wait_tensorcnt(0);
#else
  asm volatile("s_wait_tensorcnt 0x0" ::: "memory");
#endif
}
#endif  // USE_TDM

// ---------------------------------------------------------------------------
// 1) transpose + f32->f16 convert:  dst[b][r][c] = src[b][c][r]
// ---------------------------------------------------------------------------
__global__ void k_transpose_cvt(const float* __restrict__ src,
                                _Float16* __restrict__ dst,
                                int rows, int cols, int batch) {
  int idx = blockIdx.x * blockDim.x + threadIdx.x;
  int total = batch * rows * cols;
  if (idx >= total) return;
  int c = idx % cols;
  int r = (idx / cols) % rows;
  int b = idx / (rows * cols);
  dst[idx] = (_Float16)src[(b * cols + c) * rows + r];
}

// ---------------------------------------------------------------------------
// 2) projection: dst[h] = src(2048x128,f32) @ W[h](128x128)   (W^T f16 given)
//    transposed==0 -> dst row-major f16 [h][n][e]
//    transposed==1 -> dst f16 [h][e][n]   (vp^T, B-side for the PV gemm)
// ---------------------------------------------------------------------------
__global__ __launch_bounds__(32)
void k_proj(const float* __restrict__ src, const _Float16* __restrict__ WT,
            _Float16* __restrict__ dst, int transposed) {
  int lane = threadIdx.x;
  int h    = blockIdx.x >> 7;                // / (SEQ/16)
  int i0   = (blockIdx.x & 127) << 4;        // row-tile
  const _Float16* wt = WT + h * DIM * DIM;   // [e][k]

  v8f c[8];
#pragma unroll
  for (int et = 0; et < 8; ++et)
#pragma unroll
    for (int g = 0; g < 8; ++g) c[et][g] = 0.f;

  for (int kc = 0; kc < 4; ++kc) {
    v16h a = load_fragA_f32(src + i0 * DIM + kc * 32, DIM, lane);
#pragma unroll
    for (int et = 0; et < 8; ++et) {
      v16h b = load_fragB(wt + (et * 16) * DIM + kc * 32, DIM, lane);
      c[et] = wmma16(a, b, c[et]);
    }
  }

  int h2 = lane >> 4, n = lane & 15;
#pragma unroll
  for (int et = 0; et < 8; ++et)
#pragma unroll
    for (int g = 0; g < 8; ++g) {
      int row = i0 + g + 8 * h2;
      int col = et * 16 + n;
      _Float16 val = (_Float16)c[et][g];
      if (!transposed) dst[((size_t)h * SEQ + row) * DIM + col] = val;
      else             dst[((size_t)h * DIM + col) * SEQ + row] = val;
    }
}

// ---------------------------------------------------------------------------
// 3) flash attention per (head, 16-row kp tile):
//    S = kp_tile . qp^T ; masked scale ; online softmax over j ; O += P . vp
//    qp / vp^T tiles staged global->LDS by the Tensor Data Mover,
//    double-buffered so DMA of tile j+1 overlaps compute of tile j.
// ---------------------------------------------------------------------------
__global__ __launch_bounds__(32)
void k_flash(const _Float16* __restrict__ kp, const _Float16* __restrict__ qp,
             const _Float16* __restrict__ vpT, const int* __restrict__ mask,
             _Float16* __restrict__ oc) {
#if USE_TDM
  __shared__ __align__(16) _Float16 qbuf[2][32 * DIM];   // 2 x 8KB
  __shared__ __align__(16) _Float16 vbuf[2][DIM * 32];   // 2 x 8KB
#endif
  __shared__ __align__(16) _Float16 pbuf[16 * 32];       // one wave per block

  int lane = threadIdx.x;
  int h    = blockIdx.x >> 7;
  int i0   = (blockIdx.x & 127) << 4;

  const _Float16* kpb = kp  + ((size_t)h * SEQ + i0) * DIM;
  const _Float16* qpb = qp  + (size_t)h * SEQ * DIM;
  const _Float16* vtb = vpT + (size_t)h * DIM * SEQ;

  // kp tile A-fragments are loop-invariant: preload all K chunks.
  v16h afr[4];
#pragma unroll
  for (int kc = 0; kc < 4; ++kc) afr[kc] = load_fragA(kpb + kc * 32, DIM, lane);

  float mrun[8], lrun[8];
  v8f o[8];
#pragma unroll
  for (int g = 0; g < 8; ++g) { mrun[g] = -3.0e38f; lrun[g] = 0.f; }
#pragma unroll
  for (int dt = 0; dt < 8; ++dt)
#pragma unroll
    for (int g = 0; g < 8; ++g) o[dt][g] = 0.f;

  int h2 = lane >> 4, n = lane & 15;

#if USE_TDM
  // Prologue: DMA first j-tile into buffer 0.
  //  qp tile: 32 rows x 128 cols, contiguous -> 1D tile of 4096 halves.
  //  vp tile: 128 rows x 32 cols out of the [128][2048] vp^T matrix.
  tdm_load_f16(&qbuf[0][0], qpb, 32 * DIM, 0, 32 * DIM, 0, 32 * DIM);
  tdm_load_f16(&vbuf[0][0], vtb, SEQ, DIM, 32, DIM, SEQ);
  tdm_wait0();
  int cur = 0;
#endif

  for (int j0 = 0; j0 < SEQ; j0 += 32) {
#if USE_TDM
    if (j0 + 32 < SEQ) {     // kick DMA for next tile into the other buffer
      tdm_load_f16(&qbuf[cur ^ 1][0], qpb + (size_t)(j0 + 32) * DIM,
                   32 * DIM, 0, 32 * DIM, 0, 32 * DIM);
      tdm_load_f16(&vbuf[cur ^ 1][0], vtb + (j0 + 32),
                   SEQ, DIM, 32, DIM, SEQ);
    }
    const _Float16* qsrc = &qbuf[cur][0];   // [32][128], ld = DIM
    const _Float16* vsrc = &vbuf[cur][0];   // [128][32], ld = 32
    const int qld = DIM, vld = 32;
#else
    const _Float16* qsrc = qpb + (size_t)j0 * DIM;
    const _Float16* vsrc = vtb + j0;
    const int qld = DIM, vld = SEQ;
#endif
    if (j0 + 32 < SEQ)       // warm L2 for the next mask rows
      __builtin_prefetch(mask + (size_t)(i0 + ((lane >> 2) & 15)) * SEQ + j0 + 32, 0, 1);

    // ---- S = kp . qp^T for 32 columns (two 16-wide subtiles) -------------
    v8f s0, s1;
#pragma unroll
    for (int g = 0; g < 8; ++g) { s0[g] = 0.f; s1[g] = 0.f; }
#pragma unroll
    for (int kc = 0; kc < 4; ++kc) {
      v16h b0 = load_fragB(qsrc + kc * 32, qld, lane);
      v16h b1 = load_fragB(qsrc + 16 * qld + kc * 32, qld, lane);
      s0 = wmma16(afr[kc], b0, s0);
      s1 = wmma16(afr[kc], b1, s1);
    }

    // ---- mask + scale -----------------------------------------------------
    float v0[8], v1[8];
#pragma unroll
    for (int g = 0; g < 8; ++g) {
      int row = i0 + g + 8 * h2;
      const int* mr = mask + (size_t)row * SEQ + j0;
      v0[g] = mr[n]      ? s0[g] * SCALE : NEGV;
      v1[g] = mr[16 + n] ? s1[g] * SCALE : NEGV;
    }

    // ---- online softmax (per output row, reduced across half-wave) -------
#pragma unroll
    for (int g = 0; g < 8; ++g) {
      float tm    = red_max16(fmaxf(v0[g], v1[g]));
      float mnew  = fmaxf(mrun[g], tm);
      float alpha = __expf(mrun[g] - mnew);
      float p0    = __expf(v0[g] - mnew);
      float p1    = __expf(v1[g] - mnew);
      float rs    = red_sum16(p0 + p1);
      lrun[g] = lrun[g] * alpha + rs;
      mrun[g] = mnew;
#pragma unroll
      for (int dt = 0; dt < 8; ++dt) o[dt][g] *= alpha;

      int m = g + 8 * h2;                     // C-layout -> LDS tile [16][32]
      pbuf[m * 32 + n]      = (_Float16)p0;
      pbuf[m * 32 + 16 + n] = (_Float16)p1;
    }

    asm volatile("s_wait_dscnt 0x0" ::: "memory");

    // ---- O += P . vp  (P re-read from LDS in A-layout) --------------------
    v16h pf = load_fragA(pbuf, 32, lane);
#pragma unroll
    for (int dt = 0; dt < 8; ++dt) {
      v16h bv = load_fragB(vsrc + (size_t)(dt * 16) * vld, vld, lane);
      o[dt] = wmma16(pf, bv, o[dt]);
    }

#if USE_TDM
    if (j0 + 32 < SEQ) tdm_wait0();           // next tile resident before reuse
    cur ^= 1;
#endif
  }

  // ---- epilogue: normalize and emit concatenated-head f16 [n][h*128+d] ----
#pragma unroll
  for (int dt = 0; dt < 8; ++dt)
#pragma unroll
    for (int g = 0; g < 8; ++g) {
      int row = i0 + g + 8 * h2;
      int col = h * DIM + dt * 16 + n;
      oc[(size_t)row * (HEADS * DIM) + col] = (_Float16)(o[dt][g] / lrun[g]);
    }
}

// ---------------------------------------------------------------------------
// 4) final projection: out(2048x128,f32) = oc(2048x2048,f16) @ last(2048x128)
// ---------------------------------------------------------------------------
__global__ __launch_bounds__(32)
void k_final(const _Float16* __restrict__ oc, const _Float16* __restrict__ LT,
             float* __restrict__ out) {
  int lane = threadIdx.x;
  int e0 = (blockIdx.x & 7) << 4;
  int i0 = (blockIdx.x >> 3) << 4;
  const int K = HEADS * DIM;                  // 2048

  v8f c;
#pragma unroll
  for (int g = 0; g < 8; ++g) c[g] = 0.f;

  for (int k0 = 0; k0 < K; k0 += 32) {
    v16h a = load_fragA(oc + (size_t)i0 * K + k0, K, lane);
    v16h b = load_fragB(LT + (size_t)e0 * K + k0, K, lane);
    c = wmma16(a, b, c);
  }

  int h2 = lane >> 4, n = lane & 15;
#pragma unroll
  for (int g = 0; g < 8; ++g)
    out[(size_t)(i0 + g + 8 * h2) * DIM + e0 + n] = c[g];
}

// ---------------------------------------------------------------------------
extern "C" void kernel_launch(void* const* d_in, const int* in_sizes, int n_in,
                              void* d_out, int out_size, void* d_ws, size_t ws_size,
                              hipStream_t stream) {
  const float* q    = (const float*)d_in[0];
  const float* k    = (const float*)d_in[1];
  const float* v    = (const float*)d_in[2];
  const int*   mask = (const int*)  d_in[3];
  const float* Qw   = (const float*)d_in[4];
  const float* Kw   = (const float*)d_in[5];
  const float* Vw   = (const float*)d_in[6];
  const float* last = (const float*)d_in[7];
  float* out = (float*)d_out;

  // workspace carve-up (all f16)
  _Float16* QT  = (_Float16*)d_ws;                       // [h][e][k] 16*128*128
  _Float16* KT  = QT  + (size_t)HEADS * DIM * DIM;
  _Float16* VT  = KT  + (size_t)HEADS * DIM * DIM;
  _Float16* LT  = VT  + (size_t)HEADS * DIM * DIM;       // [e][hk]   128*2048
  _Float16* qpb = LT  + (size_t)DIM * HEADS * DIM;       // [h][n][e] 16*2048*128
  _Float16* kpb = qpb + (size_t)HEADS * SEQ * DIM;
  _Float16* vtb = kpb + (size_t)HEADS * SEQ * DIM;       // [h][e][n]
  _Float16* oc  = vtb + (size_t)HEADS * SEQ * DIM;       // [n][h*e]  2048*2048

  // 1) weight / last transposes into B^T f16 layout
  int tw = HEADS * DIM * DIM;
  k_transpose_cvt<<<(tw + 255) / 256, 256, 0, stream>>>(Qw, QT, DIM, DIM, HEADS);
  k_transpose_cvt<<<(tw + 255) / 256, 256, 0, stream>>>(Kw, KT, DIM, DIM, HEADS);
  k_transpose_cvt<<<(tw + 255) / 256, 256, 0, stream>>>(Vw, VT, DIM, DIM, HEADS);
  int tl = DIM * HEADS * DIM;
  k_transpose_cvt<<<(tl + 255) / 256, 256, 0, stream>>>(last, LT, DIM, HEADS * DIM, 1);

  // 2) per-head projections
  dim3 pg(HEADS * (SEQ / 16));
  k_proj<<<pg, 32, 0, stream>>>(q, QT, qpb, 0);
  k_proj<<<pg, 32, 0, stream>>>(k, KT, kpb, 0);
  k_proj<<<pg, 32, 0, stream>>>(v, VT, vtb, 1);

  // 3) fused masked-softmax attention (flash style, never materializes att)
  k_flash<<<pg, 32, 0, stream>>>(kpb, qpb, vtb, mask, oc);

  // 4) output projection
  k_final<<<(SEQ / 16) * (DIM / 16), 32, 0, stream>>>(oc, LT, out);
}
